// LuminaT2IAdapterBlock_29240137351555
// MI455X (gfx1250) — matrix operand order repaired and checked by reading, not compiled, not run on hardware
//
#include <hip/hip_runtime.h>

typedef __attribute__((ext_vector_type(16))) __bf16 v16bf;
typedef __attribute__((ext_vector_type(8)))  float  v8f;
typedef __attribute__((ext_vector_type(4)))  int    i32x4;

constexpr int DIMN     = 2048;
constexpr int HEADS    = 16;
constexpr int KV_HEADS = 8;
constexpr int HD       = 128;
constexpr int RANKN    = 16;
constexpr int INNERN   = 5632;
constexpr int BB       = 2;
constexpr int SSEQ     = 2048;
constexpr int ROWS     = BB * SSEQ;   // 4096
constexpr float EPSF   = 1e-5f;

// ---------------- workspace arena (bytes) ----------------
constexpr size_t SMALL  = (size_t)ROWS * RANKN * 4;                  // 256 KB
constexpr size_t OFF_NH = 0;                                         // 32 MB (reused as nf)
constexpr size_t OFF_TQ = (size_t)ROWS * DIMN * 4;                   // 33554432
constexpr size_t OFF_TK = OFF_TQ + SMALL;
constexpr size_t OFF_TV = OFF_TK + SMALL;
constexpr size_t OFF_T1 = OFF_TV + SMALL;
constexpr size_t OFF_T3 = OFF_T1 + SMALL;
constexpr size_t OFF_T2 = OFF_T3 + SMALL;
constexpr size_t OFF_Q  = OFF_T2 + SMALL;                            // 35127296
constexpr size_t OFF_K  = OFF_Q  + (size_t)ROWS * DIMN * 4;
constexpr size_t OFF_V  = OFF_K  + (size_t)ROWS * KV_HEADS * HD * 4;
constexpr size_t OFF_QB = OFF_V  + (size_t)ROWS * KV_HEADS * HD * 4;
constexpr size_t OFF_KB = OFF_QB + (size_t)ROWS * DIMN * 2;
constexpr size_t OFF_VB = OFF_KB + (size_t)ROWS * KV_HEADS * HD * 2;
constexpr size_t OFF_H  = OFF_VB + (size_t)ROWS * KV_HEADS * HD * 2;
constexpr size_t OFF_G  = OFF_Q;  // 92 MB SwiGLU buffer overlays dead q/k/v/q_bf/k_bf

#if __has_builtin(__builtin_amdgcn_global_load_async_to_lds_b128)
#define USE_ASYNC_LDS 1
#else
#define USE_ASYNC_LDS 0
#endif

__device__ __forceinline__ void wait_async_zero() {
#if __has_builtin(__builtin_amdgcn_s_wait_asynccnt)
    __builtin_amdgcn_s_wait_asynccnt(0);
#else
    asm volatile("s_wait_asynccnt 0x0" ::: "memory");
#endif
}

__device__ __forceinline__ unsigned short f32_to_bf16_bits(float f) {
    union { __bf16 h; unsigned short s; } u; u.h = (__bf16)f; return u.s;
}

union AFrag { v16bf v; unsigned u[8]; };

// ---------------- RMSNorm: y = x * rsqrt(mean(x^2)+eps) * w ----------------
__global__ __launch_bounds__(256) void rms_kernel(const float* __restrict__ x,
                                                  const float* __restrict__ w,
                                                  float* __restrict__ y, int D) {
    int row = blockIdx.x, tid = threadIdx.x;
    const float* xr = x + (size_t)row * D;
    float ss = 0.f;
    for (int d = tid; d < D; d += 256) { float v = xr[d]; ss += v * v; }
    __shared__ float red[256];
    red[tid] = ss; __syncthreads();
    for (int st = 128; st > 0; st >>= 1) {
        if (tid < st) red[tid] += red[tid + st];
        __syncthreads();
    }
    float rs = rsqrtf(red[0] / (float)D + EPSF);
    float* yr = y + (size_t)row * D;
    for (int d = tid; d < D; d += 256) yr[d] = xr[d] * rs * w[d];
}

// ---------------- LayerNorm in place ----------------
__global__ __launch_bounds__(256) void ln_kernel(float* __restrict__ y,
                                                 const float* __restrict__ w,
                                                 const float* __restrict__ b, int N) {
    int row = blockIdx.x, tid = threadIdx.x;
    float* yr = y + (size_t)row * N;
    float s1 = 0.f, s2 = 0.f;
    for (int d = tid; d < N; d += 256) { float v = yr[d]; s1 += v; s2 += v * v; }
    __shared__ float r1[256], r2[256];
    r1[tid] = s1; r2[tid] = s2; __syncthreads();
    for (int st = 128; st > 0; st >>= 1) {
        if (tid < st) { r1[tid] += r1[tid + st]; r2[tid] += r2[tid + st]; }
        __syncthreads();
    }
    float mean = r1[0] / (float)N;
    float var  = r2[0] / (float)N - mean * mean;
    float rs = rsqrtf(var + EPSF);
    for (int d = tid; d < N; d += 256) yr[d] = (yr[d] - mean) * rs * w[d] + b[d];
}

// ---------------- LoRA-A: t[row][r] = sum_d x[row][d] * wa[d][r] (RANK=16) ----------------
__global__ __launch_bounds__(256) void lora_a_kernel(const float* __restrict__ x,
                                                     const float* __restrict__ wa,
                                                     float* __restrict__ t, int D) {
    int row = blockIdx.x, tid = threadIdx.x;
    int lane = tid & 31, wave = tid >> 5;
    int r = wave * 2 + (lane >> 4);
    int l16 = lane & 15;
    const float* xr = x + (size_t)row * D;
    float acc = 0.f;
    for (int d = l16; d < D; d += 16) acc += xr[d] * wa[(size_t)d * RANKN + r];
    for (int dm = 1; dm < 16; dm <<= 1) acc += __shfl_xor(acc, dm, 32);
    if (l16 == 0) t[(size_t)row * RANKN + r] = acc;
}

// ---------------- LoRA-B: y[row][j] = dot(t[row], wb[:,j])/16 + bias[j] ----------------
__global__ __launch_bounds__(256) void lora_b_kernel(const float* __restrict__ t,
                                                     const float* __restrict__ wb,
                                                     const float* __restrict__ bias,
                                                     float* __restrict__ y, int N) {
    int row = blockIdx.x;
    int j = blockIdx.y * 256 + threadIdx.x;
    if (j >= N) return;
    const float* tr = t + (size_t)row * RANKN;
    float acc = 0.f;
#pragma unroll
    for (int r = 0; r < RANKN; ++r) acc += tr[r] * wb[(size_t)r * N + j];
    y[(size_t)row * N + j] = acc * (1.f / 16.f) + bias[j];
}

// ---------------- RoPE + fp32 -> bf16 ----------------
__global__ __launch_bounds__(256) void rope_bf16_kernel(const float* __restrict__ x,
                                                        const float* __restrict__ ct,
                                                        const float* __restrict__ st,
                                                        __bf16* __restrict__ y, int heads) {
    int idx = blockIdx.x * 256 + threadIdx.x;           // ROWS*heads*64 total
    int i   = idx & 63;
    int t   = idx >> 6;
    int hh  = t % heads;
    int row = t / heads;                                // b*S + s
    int s   = row & (SSEQ - 1);
    size_t base = ((size_t)row * heads + hh) * HD + 2 * i;
    float x0 = x[base], x1 = x[base + 1];
    float c = ct[s * 64 + i], sn = st[s * 64 + i];
    y[base]     = (__bf16)(x0 * c - x1 * sn);
    y[base + 1] = (__bf16)(x0 * sn + x1 * c);
}

__global__ __launch_bounds__(256) void cvt_bf16_kernel(const float* __restrict__ x,
                                                       __bf16* __restrict__ y, int n) {
    int i = blockIdx.x * 256 + threadIdx.x;
    if (i < n) y[i] = (__bf16)x[i];
}

// ---------------- Flash attention (bf16 WMMA, fp32 accum) + residual add ----------------
// grid (S/128, HEADS, B); 256 threads = 8 waves; wave handles one 16-row Q tile.
// Double-buffered LDS; K staged global->LDS via async DMA (ASYNCcnt), V register-
// prefetched and stored transposed for the P@V B-operand.
__global__ __launch_bounds__(256) void attn_kernel(const __bf16* __restrict__ qb,
                                                   const __bf16* __restrict__ kb,
                                                   const __bf16* __restrict__ vb,
                                                   const unsigned char* __restrict__ mask,
                                                   const float* __restrict__ residual,
                                                   float* __restrict__ hout) {
    const int b = blockIdx.z, head = blockIdx.y;
    const int kvh = head >> 1;                      // n_rep = 2
    const int tid = threadIdx.x;
    const int wave = tid >> 5, lane = tid & 31;
    const int n  = lane & 15;                       // N-index (B/C cols, A row)
    const int kg = lane >> 4;                       // lane-group half
    const int qrow0 = blockIdx.x * 128 + wave * 16;

    __shared__ unsigned short Ks[2][32][HD];        // [buf][key_local][dim]   16 KB
    __shared__ unsigned short Vt[2][HD][32];        // [buf][dim][key_local]   16 KB
    __shared__ unsigned short Psh[8][16][32];       // per-wave P staging       8 KB

    // Q tile in A-layout: 4 chunks of K=32 along head_dim
    AFrag qa[4];
    {
        const unsigned short* qrow = (const unsigned short*)(qb +
            ((size_t)((b * SSEQ + qrow0 + n) * HEADS + head)) * HD);
#pragma unroll
        for (int c = 0; c < 4; ++c)
#pragma unroll
            for (int j = 0; j < 8; ++j) {
                int kd = c * 32 + ((j < 4) ? (kg * 8 + 2 * j) : (16 + kg * 8 + 2 * (j - 4)));
                qa[c].u[j] = *(const unsigned*)(qrow + kd);
            }
    }

    v8f O[8];
#pragma unroll
    for (int i = 0; i < 8; ++i)
#pragma unroll
        for (int r = 0; r < 8; ++r) O[i][r] = 0.f;
    float mrow[8], lrow[8];
#pragma unroll
    for (int r = 0; r < 8; ++r) { mrow[r] = -3.0e38f; lrow[r] = 0.f; }

    const float scale = 0.08838834764831845f;       // 1/sqrt(128)
    const size_t kvStride = (size_t)KV_HEADS * HD;  // elements between keys
    const __bf16* kBase = kb + ((size_t)(b * SSEQ) * KV_HEADS + kvh) * HD;
    const __bf16* vBase = vb + ((size_t)(b * SSEQ) * KV_HEADS + kvh) * HD;

    unsigned vreg[8];
#if !USE_ASYNC_LDS
    unsigned kreg[8];
#endif

    auto load_v_regs = [&](int kbase) {
#pragma unroll
        for (int t = 0; t < 8; ++t) {
            int i = tid + 256 * t;
            int key = i >> 6, dw = i & 63;
            vreg[t] = *((const unsigned*)(vBase + (size_t)(kbase + key) * kvStride) + dw);
        }
    };
#if USE_ASYNC_LDS
    auto issue_k_async = [&](int kbase, int dstbuf) {
#pragma unroll
        for (int t = 0; t < 2; ++t) {
            int u = tid + 256 * t;                  // 512 x b128 units
            int key = u >> 4, seg = u & 15;
            const __bf16* gp = kBase + (size_t)(kbase + key) * kvStride + seg * 8;
            __builtin_amdgcn_global_load_async_to_lds_b128(
                (__attribute__((address_space(1))) i32x4*)gp,
                (__attribute__((address_space(3))) i32x4*)&Ks[dstbuf][key][seg * 8],
                0, 0);
        }
    };
#else
    auto load_k_regs = [&](int kbase) {
#pragma unroll
        for (int t = 0; t < 8; ++t) {
            int i = tid + 256 * t;
            int key = i >> 6, dw = i & 63;
            kreg[t] = *((const unsigned*)(kBase + (size_t)(kbase + key) * kvStride) + dw);
        }
    };
#endif

    // prologue: stage chunk 0
    load_v_regs(0);
#if USE_ASYNC_LDS
    issue_k_async(0, 0);
#else
    load_k_regs(0);
#endif

    int buf = 0;
    for (int kc = 0; kc < SSEQ / 32; ++kc) {
        const int kbase = kc * 32;
#if USE_ASYNC_LDS
        wait_async_zero();                          // my async K writes into Ks[buf] done
#endif
        // V -> LDS transposed (and K in fallback path)
#pragma unroll
        for (int t = 0; t < 8; ++t) {
            int i = tid + 256 * t;
            int key = i >> 6, dw = i & 63;
            unsigned vv = vreg[t];
            Vt[buf][dw * 2][key]     = (unsigned short)(vv & 0xffffu);
            Vt[buf][dw * 2 + 1][key] = (unsigned short)(vv >> 16);
#if !USE_ASYNC_LDS
            *(unsigned*)&Ks[buf][key][dw * 2] = kreg[t];
#endif
        }
        __syncthreads();                            // all waves staged; buf^1 now free
        if (kc + 1 < SSEQ / 32) {                   // prefetch next chunk under compute
            load_v_regs(kbase + 32);
#if USE_ASYNC_LDS
            issue_k_async(kbase + 32, buf ^ 1);
#else
            load_k_regs(kbase + 32);
#endif
        }

        // S = Q x K^T : two 16x16 tiles, K-dim 128 in 4 chunks
        v8f s[2];
#pragma unroll
        for (int t = 0; t < 2; ++t) {
            v8f acc;
#pragma unroll
            for (int r = 0; r < 8; ++r) acc[r] = 0.f;
#pragma unroll
            for (int c = 0; c < 4; ++c) {
                AFrag bk_;
#pragma unroll
                for (int j = 0; j < 8; ++j)
                    bk_.u[j] = *(const unsigned*)&Ks[buf][t * 16 + n][c * 32 + ((j < 4) ? (kg * 8 + 2 * j) : (16 + kg * 8 + 2 * (j - 4)))];
                acc = __builtin_amdgcn_wmma_f32_16x16x32_bf16(false, qa[c].v, false, bk_.v,
                                                              (short)0, acc, false, false);
            }
            s[t] = acc;
        }

        // scale + mask
        const bool mk0 = mask[b * SSEQ + kbase + n] != 0;
        const bool mk1 = mask[b * SSEQ + kbase + 16 + n] != 0;
#pragma unroll
        for (int r = 0; r < 8; ++r) {
            float v0 = s[0][r] * scale; if (!mk0) v0 = -3.0e38f;
            float v1 = s[1][r] * scale; if (!mk1) v1 = -3.0e38f;
            s[0][r] = v0; s[1][r] = v1;
        }

        // online softmax (row reductions across the 16-lane group)
        float alpha[8];
#pragma unroll
        for (int r = 0; r < 8; ++r) {
            float mx = fmaxf(s[0][r], s[1][r]);
#pragma unroll
            for (int dm = 1; dm < 16; dm <<= 1) mx = fmaxf(mx, __shfl_xor(mx, dm, 32));
            float mnew = fmaxf(mrow[r], mx);
            float p0 = __expf(s[0][r] - mnew);
            float p1 = __expf(s[1][r] - mnew);
            s[0][r] = p0; s[1][r] = p1;
            float ps = p0 + p1;
#pragma unroll
            for (int dm = 1; dm < 16; dm <<= 1) ps += __shfl_xor(ps, dm, 32);
            alpha[r] = __expf(mrow[r] - mnew);
            lrow[r] = lrow[r] * alpha[r] + ps;
            mrow[r] = mnew;
        }
#pragma unroll
        for (int nc = 0; nc < 8; ++nc)
#pragma unroll
            for (int r = 0; r < 8; ++r) O[nc][r] *= alpha[r];

        // P: C-layout -> bf16 LDS -> A-layout (wave-private region)
#pragma unroll
        for (int t = 0; t < 2; ++t)
#pragma unroll
            for (int r = 0; r < 8; ++r)
                Psh[wave][r + 8 * kg][t * 16 + n] = f32_to_bf16_bits(s[t][r]);
        __builtin_amdgcn_wave_barrier();
        AFrag pa;
#pragma unroll
        for (int j = 0; j < 8; ++j) {
            int kk = (j < 4) ? (kg * 8 + 2 * j) : (16 + kg * 8 + 2 * (j - 4));
            pa.u[j] = *(const unsigned*)&Psh[wave][n][kk];
        }

        // O += P x V : 8 output chunks of 16 dims, K-dim = 32 keys
#pragma unroll
        for (int nc = 0; nc < 8; ++nc) {
            AFrag vB;
#pragma unroll
            for (int j = 0; j < 8; ++j)
                vB.u[j] = *(const unsigned*)&Vt[buf][nc * 16 + n][(j < 4) ? (kg * 8 + 2 * j) : (16 + kg * 8 + 2 * (j - 4))];
            O[nc] = __builtin_amdgcn_wmma_f32_16x16x32_bf16(false, pa.v, false, vB.v,
                                                            (short)0, O[nc], false, false);
        }
        buf ^= 1;
    }

    // normalize + fused residual add: h = residual + attn_out
#pragma unroll
    for (int r = 0; r < 8; ++r) {
        const float inv = 1.0f / lrow[r];
        const int row = qrow0 + r + 8 * kg;
#pragma unroll
        for (int nc = 0; nc < 8; ++nc) {
            size_t idx = (size_t)(b * SSEQ + row) * DIMN + head * HD + nc * 16 + n;
            hout[idx] = residual[idx] + O[nc][r] * inv;
        }
    }
}

// ---------------- SwiGLU mid: g = silu(t1@w1b/16+b1) * (t3@w3b/16+b3) ----------------
__global__ __launch_bounds__(256) void ffn_mid_kernel(const float* __restrict__ t1,
                                                      const float* __restrict__ t3,
                                                      const float* __restrict__ w1b,
                                                      const float* __restrict__ b1,
                                                      const float* __restrict__ w3b,
                                                      const float* __restrict__ b3,
                                                      float* __restrict__ g) {
    int row = blockIdx.x;
    int j = blockIdx.y * 256 + threadIdx.x;
    if (j >= INNERN) return;
    const float* a1 = t1 + (size_t)row * RANKN;
    const float* a3 = t3 + (size_t)row * RANKN;
    float u1 = 0.f, u3 = 0.f;
#pragma unroll
    for (int r = 0; r < RANKN; ++r) {
        u1 += a1[r] * w1b[(size_t)r * INNERN + j];
        u3 += a3[r] * w3b[(size_t)r * INNERN + j];
    }
    u1 = u1 * (1.f / 16.f) + b1[j];
    u3 = u3 * (1.f / 16.f) + b3[j];
    float sig = 1.f / (1.f + __expf(-u1));
    g[(size_t)row * INNERN + j] = (u1 * sig) * u3;
}

// ---------------- out = h + t2@w2b/16 + b2 ----------------
__global__ __launch_bounds__(256) void final_kernel(const float* __restrict__ h,
                                                    const float* __restrict__ t2,
                                                    const float* __restrict__ w2b,
                                                    const float* __restrict__ b2,
                                                    float* __restrict__ out) {
    int row = blockIdx.x;
    int j = blockIdx.y * 256 + threadIdx.x;
    const float* tr = t2 + (size_t)row * RANKN;
    float acc = 0.f;
#pragma unroll
    for (int r = 0; r < RANKN; ++r) acc += tr[r] * w2b[(size_t)r * DIMN + j];
    size_t idx = (size_t)row * DIMN + j;
    out[idx] = h[idx] + acc * (1.f / 16.f) + b2[j];
}

extern "C" void kernel_launch(void* const* d_in, const int* in_sizes, int n_in,
                              void* d_out, int out_size, void* d_ws, size_t ws_size,
                              hipStream_t stream) {
    (void)in_sizes; (void)n_in; (void)out_size; (void)ws_size;
    const float* hid   = (const float*)d_in[0];
    const unsigned char* mask = (const unsigned char*)d_in[1];
    const float* cosb  = (const float*)d_in[2];
    const float* sinb  = (const float*)d_in[3];
    const float* wq_a  = (const float*)d_in[4];
    const float* wq_b  = (const float*)d_in[5];
    const float* bq    = (const float*)d_in[6];
    const float* wk_a  = (const float*)d_in[7];
    const float* wk_b  = (const float*)d_in[8];
    const float* bk    = (const float*)d_in[9];
    const float* wv_a  = (const float*)d_in[10];
    const float* wv_b  = (const float*)d_in[11];
    const float* bv    = (const float*)d_in[12];
    const float* nqw   = (const float*)d_in[13];
    const float* nqb   = (const float*)d_in[14];
    const float* nkw   = (const float*)d_in[15];
    const float* nkb   = (const float*)d_in[16];
    const float* w1a   = (const float*)d_in[17];
    const float* w1b   = (const float*)d_in[18];
    const float* b1    = (const float*)d_in[19];
    const float* w3a   = (const float*)d_in[20];
    const float* w3b   = (const float*)d_in[21];
    const float* b3    = (const float*)d_in[22];
    const float* w2a   = (const float*)d_in[23];
    const float* w2b   = (const float*)d_in[24];
    const float* b2    = (const float*)d_in[25];
    const float* normw = (const float*)d_in[26];
    const float* ffnw  = (const float*)d_in[27];

    char* ws = (char*)d_ws;
    float*  nh = (float*)(ws + OFF_NH);
    float*  tq = (float*)(ws + OFF_TQ);
    float*  tk = (float*)(ws + OFF_TK);
    float*  tv = (float*)(ws + OFF_TV);
    float*  t1 = (float*)(ws + OFF_T1);
    float*  t3 = (float*)(ws + OFF_T3);
    float*  t2 = (float*)(ws + OFF_T2);
    float*  q  = (float*)(ws + OFF_Q);
    float*  k  = (float*)(ws + OFF_K);
    float*  v  = (float*)(ws + OFF_V);
    __bf16* qbf = (__bf16*)(ws + OFF_QB);
    __bf16* kbf = (__bf16*)(ws + OFF_KB);
    __bf16* vbf = (__bf16*)(ws + OFF_VB);
    float*  h  = (float*)(ws + OFF_H);
    float*  g  = (float*)(ws + OFF_G);
    float*  nf = nh;                       // reuse
    float*  out = (float*)d_out;

    // attention input path
    rms_kernel<<<ROWS, 256, 0, stream>>>(hid, normw, nh, DIMN);
    lora_a_kernel<<<ROWS, 256, 0, stream>>>(nh, wq_a, tq, DIMN);
    lora_a_kernel<<<ROWS, 256, 0, stream>>>(nh, wk_a, tk, DIMN);
    lora_a_kernel<<<ROWS, 256, 0, stream>>>(nh, wv_a, tv, DIMN);
    lora_b_kernel<<<dim3(ROWS, DIMN / 256), 256, 0, stream>>>(tq, wq_b, bq, q, DIMN);
    lora_b_kernel<<<dim3(ROWS, (KV_HEADS * HD) / 256), 256, 0, stream>>>(tk, wk_b, bk, k, KV_HEADS * HD);
    lora_b_kernel<<<dim3(ROWS, (KV_HEADS * HD) / 256), 256, 0, stream>>>(tv, wv_b, bv, v, KV_HEADS * HD);
    ln_kernel<<<ROWS, 256, 0, stream>>>(q, nqw, nqb, DIMN);
    ln_kernel<<<ROWS, 256, 0, stream>>>(k, nkw, nkb, KV_HEADS * HD);
    rope_bf16_kernel<<<(ROWS * HEADS * 64) / 256, 256, 0, stream>>>(q, cosb, sinb, qbf, HEADS);
    rope_bf16_kernel<<<(ROWS * KV_HEADS * 64) / 256, 256, 0, stream>>>(k, cosb, sinb, kbf, KV_HEADS);
    cvt_bf16_kernel<<<(ROWS * KV_HEADS * HD) / 256, 256, 0, stream>>>(v, vbf, ROWS * KV_HEADS * HD);

    // flash attention + residual
    attn_kernel<<<dim3(SSEQ / 128, HEADS, BB), 256, 0, stream>>>(qbf, kbf, vbf, mask, hid, h);

    // FFN path
    rms_kernel<<<ROWS, 256, 0, stream>>>(h, ffnw, nf, DIMN);
    lora_a_kernel<<<ROWS, 256, 0, stream>>>(nf, w1a, t1, DIMN);
    lora_a_kernel<<<ROWS, 256, 0, stream>>>(nf, w3a, t3, DIMN);
    ffn_mid_kernel<<<dim3(ROWS, INNERN / 256), 256, 0, stream>>>(t1, t3, w1b, b1, w3b, b3, g);
    lora_a_kernel<<<ROWS, 256, 0, stream>>>(g, w2a, t2, INNERN);
    final_kernel<<<dim3(ROWS, DIMN / 256), 256, 0, stream>>>(h, t2, w2b, b2, out);
}
